// GPS_20306605376072
// MI455X (gfx1250) — matrix operand order repaired and checked by reading, not compiled, not run on hardware
//
#include <hip/hip_runtime.h>

// ---------------------------------------------------------------------------
// GraphGPS forward for MI455X (gfx1250, wave32, WMMA).
// All dense matmuls use v_wmma_f32_16x16x32_bf16 (bf16 in, f32 accumulate).
// Each GEMM wave owns a 16x128 output stripe: A packed once per k-step and
// reused across 8 consecutive WMMAs (8x less activation traffic).
// GINE aggregation uses per-graph LDS + ds_add_f32 atomics.
// Attention is one fused kernel per (graph, head) with LDS-staged Q/K/V/P.
// ---------------------------------------------------------------------------

typedef __attribute__((ext_vector_type(16))) __bf16 v16bf;
typedef __attribute__((ext_vector_type(8)))  __bf16 v8bf;
typedef __attribute__((ext_vector_type(8)))  float  v8f;

#define NTOT 65536   // total nodes (G*N)
#define CDIM 128     // channels
#define GNUM 512     // graphs
#define NPG  128     // nodes per graph
#define ETOT 524288  // total edges
#define EPG  1024    // edges per graph
#define NLAY 3
#define BN_EPS 1e-5f
#define INV_NT (1.0f / 65536.0f)
#define INV_SQRT_HD 0.17677669529663687f

#define WMMA_BF16(a, b, c) \
  __builtin_amdgcn_wmma_f32_16x16x32_bf16(false, (a), false, (b), (short)0, (c), false, false)

// Pack 16 f32 -> v16bf in the WMMA A-operand per-lane layout:
// elements 0..7  = rowptr[kb .. kb+7]
// elements 8..15 = rowptr[kb+16 .. kb+23]
static __device__ __forceinline__ v16bf pack_a_f32(const float* __restrict__ rowptr, int kb) {
  const float4* p0 = reinterpret_cast<const float4*>(rowptr + kb);
  const float4* p1 = reinterpret_cast<const float4*>(rowptr + kb + 16);
  float4 a0 = p0[0], a1 = p0[1], a2 = p1[0], a3 = p1[1];
  v16bf r;
  r[0]  = (__bf16)a0.x; r[1]  = (__bf16)a0.y; r[2]  = (__bf16)a0.z; r[3]  = (__bf16)a0.w;
  r[4]  = (__bf16)a1.x; r[5]  = (__bf16)a1.y; r[6]  = (__bf16)a1.z; r[7]  = (__bf16)a1.w;
  r[8]  = (__bf16)a2.x; r[9]  = (__bf16)a2.y; r[10] = (__bf16)a2.z; r[11] = (__bf16)a2.w;
  r[12] = (__bf16)a3.x; r[13] = (__bf16)a3.y; r[14] = (__bf16)a3.z; r[15] = (__bf16)a3.w;
  return r;
}

static __device__ __forceinline__ v16bf pack_a_lds(const __bf16* __restrict__ p, int kb) {
  v8bf lo = *reinterpret_cast<const v8bf*>(p + kb);
  v8bf hi = *reinterpret_cast<const v8bf*>(p + kb + 8);
  return __builtin_shufflevector(lo, hi, 0,1,2,3,4,5,6,7,8,9,10,11,12,13,14,15);
}

// ---------------------------------------------------------------------------
// Weight pre-conversion: f32 [K,Nc] (or transposed [Nc,K]) -> bf16 in WMMA
// B-operand order: dst[((s*ntn+t)*32+lane)*16+e] = B[s*32+8*(lane>=16)+(e&7)+16*(e>=8)][t*16+(lane&15)]
// ---------------------------------------------------------------------------
__global__ void conv_weight_kernel(const float* __restrict__ src, __bf16* __restrict__ dst,
                                   int K, int Nc, int transposed) {
  int o = blockIdx.x * 256 + threadIdx.x;
  if (o >= K * Nc) return;
  int e  = o & 15;
  int l  = (o >> 4) & 31;
  int st = o >> 9;
  int ntn = Nc >> 4;
  int t = st % ntn;
  int s = st / ntn;
  int k = s * 32 + ((l >= 16) ? 8 : 0) + (e & 7) + ((e >= 8) ? 16 : 0);
  int n = t * 16 + (l & 15);
  float v = transposed ? src[(size_t)n * K + k] : src[(size_t)k * Nc + n];
  dst[o] = (__bf16)v;
}

// ---------------------------------------------------------------------------
// Fused WMMA GEMM: out[M,Nc] = epilogue(A[M,K] @ W[K,Nc] + bias)
// One wave owns a 16-row x 128-col stripe = 8 accumulator tiles.
// A operand is packed once per k-step and reused across 8 WMMAs.
// grid: (M/16) * (Nc/128) waves, 8 waves/block.
// ---------------------------------------------------------------------------
__global__ void __launch_bounds__(256)
gemm_bf16_wmma(const float* __restrict__ A, int lda,
               const __bf16* __restrict__ Wop,
               const float* __restrict__ bias,
               const float* __restrict__ res,
               float* __restrict__ out,
               int Nc, int K, int do_relu) {
  const int lane = threadIdx.x & 31;
  const int wid  = blockIdx.x * 8 + (threadIdx.x >> 5);
  const int nstripes = Nc >> 7;            // column stripes of 128
  const int rt = wid / nstripes;           // row tile
  const int cs = wid % nstripes;           // column stripe
  const int ntn = Nc >> 4;                 // 16-wide column tiles in Nc
  const int colb = cs * 128 + (lane & 15); // this lane's column within tile 0

  v8f acc[8];
#pragma unroll
  for (int ct = 0; ct < 8; ++ct) {
    float b0 = bias ? bias[colb + ct * 16] : 0.0f;
#pragma unroll
    for (int r = 0; r < 8; ++r) acc[ct][r] = b0;
  }

  const int rowA   = rt * 16 + (lane & 15);
  const int kb_off = (lane >= 16) ? 8 : 0;
  const float* arow = A + (size_t)rowA * lda;
  const int ksteps = K >> 5;

  for (int s = 0; s < ksteps; ++s) {
    v16bf av = pack_a_f32(arow, s * 32 + kb_off);
    const __bf16* wrow = Wop + ((size_t)(s * ntn + cs * 8) * 32 + lane) * 16;
#pragma unroll
    for (int ct = 0; ct < 8; ++ct) {
      v16bf bv = *reinterpret_cast<const v16bf*>(wrow + (size_t)ct * 32 * 16);
      acc[ct] = WMMA_BF16(av, bv, acc[ct]);
    }
  }

  const int rbase = rt * 16 + ((lane < 16) ? 0 : 8);
#pragma unroll
  for (int ct = 0; ct < 8; ++ct) {
#pragma unroll
    for (int r = 0; r < 8; ++r) {
      size_t idx = (size_t)(rbase + r) * Nc + colb + ct * 16;
      float v = acc[ct][r];
      if (res) v += res[idx];
      if (do_relu) v = fmaxf(v, 0.0f);
      out[idx] = v;
    }
  }
}

// ---------------------------------------------------------------------------
// Node/PE embedding: h = x @ node_w + node_b + pe @ pe_w + pe_b
// ---------------------------------------------------------------------------
__global__ void embed_kernel(const float* __restrict__ x, const float* __restrict__ pe,
                             const float* __restrict__ nw, const float* __restrict__ nb,
                             const float* __restrict__ pw, const float* __restrict__ pb,
                             float* __restrict__ h) {
  int idx = blockIdx.x * 256 + threadIdx.x;
  int n = idx >> 7, c = idx & 127;
  float s = nb[c] + pb[c];
  const float* xr = x + (size_t)n * 9;
#pragma unroll
  for (int j = 0; j < 9; ++j) s += xr[j] * nw[j * CDIM + c];
  const float* pr = pe + (size_t)n * 20;
#pragma unroll
  for (int j = 0; j < 20; ++j) s += pr[j] * pw[j * CDIM + c];
  h[idx] = s;
}

// ---------------------------------------------------------------------------
// GINE aggregation (one block per graph): LDS accumulator 128x128 f32,
// edge embedding (K=3) recomputed in-flight, ds_add_f32 atomics, then
// z0 = h + agg.
// ---------------------------------------------------------------------------
__global__ void __launch_bounds__(256)
gine_agg_kernel(const float* __restrict__ h, const float* __restrict__ eattr,
                const float* __restrict__ ew, const float* __restrict__ eb,
                const int* __restrict__ srcI, const int* __restrict__ dstI,
                float* __restrict__ z0) {
  extern __shared__ float acc[]; // 128*128 f32 = 64 KB
  const int g = blockIdx.x;
  const int tid = threadIdx.x;
  for (int i = tid; i < NPG * CDIM; i += 256) acc[i] = 0.0f;
  __syncthreads();
  const int ebase = g * EPG;
  for (int i = tid; i < EPG * CDIM; i += 256) {
    int el = i >> 7, c = i & 127;
    int e = ebase + el;
    int s = srcI[e];
    int d = dstI[e] - g * NPG;
    float a0 = eattr[e * 3], a1 = eattr[e * 3 + 1], a2 = eattr[e * 3 + 2];
    float eaval = a0 * ew[c] + a1 * ew[CDIM + c] + a2 * ew[2 * CDIM + c] + eb[c];
    float v = h[(size_t)s * CDIM + c] + eaval;
    v = fmaxf(v, 0.0f);
    atomicAdd(&acc[d * CDIM + c], v);  // ds_add_f32
  }
  __syncthreads();
  const size_t nb = (size_t)g * NPG * CDIM;
  for (int i = tid; i < NPG * CDIM; i += 256) z0[nb + i] = h[nb + i] + acc[i];
}

// ---------------------------------------------------------------------------
// Fused attention per (graph, head): blockIdx = g*4 + head; 256 threads.
// LDS: q[128x32]bf16, k[128x32]bf16, vT[32x128]bf16, S[128x128]f32, P[128x128]bf16
// ---------------------------------------------------------------------------
__global__ void __launch_bounds__(256)
attn_kernel(const float* __restrict__ qkv, float* __restrict__ oa) {
  extern __shared__ char smem[];
  __bf16* qb  = (__bf16*)smem;            // 8 KB
  __bf16* kbm = qb + 128 * 32;            // 8 KB
  __bf16* vt  = kbm + 128 * 32;           // 8 KB (transposed: vt[hd][n])
  float*  sS  = (float*)(vt + 32 * 128);  // 64 KB
  __bf16* pb  = (__bf16*)(sS + 128 * 128);// 32 KB

  const int g   = blockIdx.x >> 2;
  const int hh  = blockIdx.x & 3;
  const int tid = threadIdx.x;

  // stage Q, K, V (V transposed) into LDS as bf16
  for (int idx = tid; idx < 128 * 32; idx += 256) {
    int n = idx >> 5, c = idx & 31;
    const float* base = qkv + ((size_t)(g * NPG + n)) * 384 + hh * 32 + c;
    qb[idx]  = (__bf16)base[0];
    kbm[idx] = (__bf16)base[128];
    vt[c * 128 + n] = (__bf16)base[256];
  }
  __syncthreads();

  const int lane = tid & 31;
  const int w    = tid >> 5;

  // S = (Q @ K^T) * 1/sqrt(HD) ; wave w owns row tile w, loops 8 column tiles
  {
    const int rowA = w * 16 + (lane & 15);
    const int kbq  = (lane >= 16) ? 16 : 0;
    v16bf av = pack_a_lds(qb + rowA * 32, kbq);
    const int kbk = (lane >= 16) ? 16 : 0;
    const int rbase = w * 16 + ((lane < 16) ? 0 : 8);
#pragma unroll
    for (int ct = 0; ct < 8; ++ct) {
      int colm = ct * 16 + (lane & 15);
      v16bf bv = *reinterpret_cast<const v16bf*>(kbm + colm * 32 + kbk);
      v8f acc = {0.f, 0.f, 0.f, 0.f, 0.f, 0.f, 0.f, 0.f};
      acc = WMMA_BF16(av, bv, acc);
#pragma unroll
      for (int r = 0; r < 8; ++r)
        sS[(rbase + r) * 128 + colm] = acc[r] * INV_SQRT_HD;
    }
  }
  __syncthreads();

  // row softmax -> P (bf16)
  if (tid < 128) {
    float* row = sS + tid * 128;
    float m = -3.4e38f;
    for (int c = 0; c < 128; ++c) m = fmaxf(m, row[c]);
    float sum = 0.0f;
    for (int c = 0; c < 128; ++c) sum += __expf(row[c] - m);
    float inv = 1.0f / sum;
    __bf16* prow = pb + tid * 128;
    for (int c = 0; c < 128; ++c) prow[c] = (__bf16)(__expf(row[c] - m) * inv);
  }
  __syncthreads();

  // O = P @ V ; wave w owns row tile w, 2 column tiles (HD=32), K=128 (4 steps)
  {
    const int rowA   = w * 16 + (lane & 15);
    const int kboffA = (lane >= 16) ? 16 : 0;
    const int kboffB = (lane >= 16) ? 16 : 0;
    const int rbase  = w * 16 + ((lane < 16) ? 0 : 8);
#pragma unroll
    for (int ct = 0; ct < 2; ++ct) {
      int hd = ct * 16 + (lane & 15);
      v8f acc = {0.f, 0.f, 0.f, 0.f, 0.f, 0.f, 0.f, 0.f};
#pragma unroll
      for (int s4 = 0; s4 < 4; ++s4) {
        v16bf av = pack_a_lds(pb + rowA * 128, s4 * 32 + kboffA);
        v16bf bv = *reinterpret_cast<const v16bf*>(vt + hd * 128 + s4 * 32 + kboffB);
        acc = WMMA_BF16(av, bv, acc);
      }
#pragma unroll
      for (int r = 0; r < 8; ++r)
        oa[((size_t)(g * NPG + rbase + r)) * CDIM + hh * 32 + hd] = acc[r];
    }
  }
}

// ---------------------------------------------------------------------------
// BatchNorm over node dimension: stats (sum, sumsq per column) then apply.
// ---------------------------------------------------------------------------
__global__ void bn_zero_kernel(float* __restrict__ stats) { stats[threadIdx.x] = 0.0f; }

__global__ void bn_stats_kernel(const float* __restrict__ x, float* __restrict__ stats) {
  int c  = threadIdx.x;       // 128 threads
  int r0 = blockIdx.x * 256;  // 256 rows per block
  float s1 = 0.0f, s2 = 0.0f;
  for (int r = 0; r < 256; ++r) {
    float v = x[(size_t)(r0 + r) * CDIM + c];
    s1 += v; s2 += v * v;
  }
  atomicAdd(&stats[c], s1);
  atomicAdd(&stats[CDIM + c], s2);
}

__global__ void bn_apply_kernel(const float* __restrict__ x, const float* __restrict__ stats,
                                const float* __restrict__ gamma, const float* __restrict__ beta,
                                const float* __restrict__ add, float* __restrict__ out) {
  int idx = blockIdx.x * 256 + threadIdx.x;
  int c = idx & 127;
  float m  = stats[c] * INV_NT;
  float vv = stats[CDIM + c] * INV_NT - m * m;
  float val = (x[idx] - m) * rsqrtf(vv + BN_EPS) * gamma[c] + beta[c];
  if (add) val += add[idx];
  out[idx] = val;
}

// ---------------------------------------------------------------------------
// Final: add-pool per graph + linear C->11
// ---------------------------------------------------------------------------
__global__ void __launch_bounds__(128)
pool_linear_kernel(const float* __restrict__ h, const float* __restrict__ lw,
                   const float* __restrict__ lb, float* __restrict__ out) {
  __shared__ float pooled[CDIM];
  const int g = blockIdx.x;
  const int c = threadIdx.x;
  float s = 0.0f;
  for (int n = 0; n < NPG; ++n) s += h[((size_t)g * NPG + n) * CDIM + c];
  pooled[c] = s;
  __syncthreads();
  if (c < 11) {
    float o = lb[c];
    for (int k = 0; k < CDIM; ++k) o += pooled[k] * lw[k * 11 + c];
    out[g * 11 + c] = o;
  }
}

// ---------------------------------------------------------------------------
extern "C" void kernel_launch(void* const* d_in, const int* in_sizes, int n_in,
                              void* d_out, int out_size, void* d_ws, size_t ws_size,
                              hipStream_t stream) {
  (void)in_sizes; (void)n_in; (void)out_size; (void)ws_size;
  const float* x    = (const float*)d_in[0];
  const float* pe   = (const float*)d_in[1];
  const float* eatt = (const float*)d_in[2];
  const int*   ei   = (const int*)d_in[3];
  const int* srcI = ei;
  const int* dstI = ei + ETOT;

  const float* node_w = (const float*)d_in[5];
  const float* node_b = (const float*)d_in[6];
  const float* pe_w   = (const float*)d_in[7];
  const float* pe_b   = (const float*)d_in[8];
  const float* edge_w = (const float*)d_in[9];
  const float* edge_b = (const float*)d_in[10];
  const float* gine_w1 = (const float*)d_in[11];
  const float* gine_b1 = (const float*)d_in[12];
  const float* gine_w2 = (const float*)d_in[13];
  const float* gine_b2 = (const float*)d_in[14];
  const float* attn_in_w  = (const float*)d_in[15];
  const float* attn_in_b  = (const float*)d_in[16];
  const float* attn_out_w = (const float*)d_in[17];
  const float* attn_out_b = (const float*)d_in[18];
  const float* mlp_w1 = (const float*)d_in[19];
  const float* mlp_b1 = (const float*)d_in[20];
  const float* mlp_w2 = (const float*)d_in[21];
  const float* mlp_b2 = (const float*)d_in[22];
  const float* g1  = (const float*)d_in[23];
  const float* be1 = (const float*)d_in[24];
  const float* g2  = (const float*)d_in[25];
  const float* be2 = (const float*)d_in[26];
  const float* g3  = (const float*)d_in[27];
  const float* be3 = (const float*)d_in[28];
  const float* lin_w = (const float*)d_in[29];
  const float* lin_b = (const float*)d_in[30];

  char* ws = (char*)d_ws;
  const size_t SZ = (size_t)NTOT * CDIM * sizeof(float); // 32 MB per [NT,C] tensor
  float* h    = (float*)(ws + 0 * SZ);
  float* z0   = (float*)(ws + 1 * SZ);
  float* t1   = (float*)(ws + 2 * SZ);
  float* r1   = (float*)(ws + 3 * SZ);
  float* h1   = (float*)(ws + 4 * SZ);
  float* oa   = (float*)(ws + 5 * SZ);
  float* r2   = (float*)(ws + 6 * SZ);
  float* o12  = (float*)(ws + 7 * SZ);
  float* r3   = (float*)(ws + 8 * SZ);
  float* qkv  = (float*)(ws + 9 * SZ);   // NT x 384 (3*SZ)
  float* t2   = (float*)(ws + 12 * SZ);  // NT x 256 (2*SZ)
  float* stats = (float*)(ws + 14 * SZ);
  __bf16* wop = (__bf16*)(ws + 14 * SZ + 4096);

  const size_t WL = 163840; // bf16 elements of converted weights per layer
  auto g1op  = [&](int i) { return wop + i * WL + 0; };
  auto g2op  = [&](int i) { return wop + i * WL + 16384; };
  auto inop  = [&](int i) { return wop + i * WL + 32768; };
  auto outop = [&](int i) { return wop + i * WL + 81920; };
  auto m1op  = [&](int i) { return wop + i * WL + 98304; };
  auto m2op  = [&](int i) { return wop + i * WL + 131072; };

  // ---- weight conversion (tiny) ----
  for (int i = 0; i < NLAY; ++i) {
    conv_weight_kernel<<<64, 256, 0, stream>>>(gine_w1 + i * 16384, g1op(i), 128, 128, 0);
    conv_weight_kernel<<<64, 256, 0, stream>>>(gine_w2 + i * 16384, g2op(i), 128, 128, 0);
    conv_weight_kernel<<<192, 256, 0, stream>>>(attn_in_w + i * 49152, inop(i), 128, 384, 1);
    conv_weight_kernel<<<64, 256, 0, stream>>>(attn_out_w + i * 16384, outop(i), 128, 128, 1);
    conv_weight_kernel<<<128, 256, 0, stream>>>(mlp_w1 + i * 32768, m1op(i), 128, 256, 0);
    conv_weight_kernel<<<128, 256, 0, stream>>>(mlp_w2 + i * 32768, m2op(i), 256, 128, 0);
  }

  // ---- embedding ----
  embed_kernel<<<NTOT * CDIM / 256, 256, 0, stream>>>(x, pe, node_w, node_b, pe_w, pe_b, h);

  const int EL_BLOCKS = NTOT * CDIM / 256;       // elementwise grid
  auto run_bn = [&](const float* xin, const float* gamma, const float* beta,
                    const float* add, float* outb) {
    bn_zero_kernel<<<1, 256, 0, stream>>>(stats);
    bn_stats_kernel<<<NTOT / 256, 128, 0, stream>>>(xin, stats);
    bn_apply_kernel<<<EL_BLOCKS, 256, 0, stream>>>(xin, stats, gamma, beta, add, outb);
  };

  // GEMM grids: (M/16) row tiles * (Nc/128) stripes waves, 8 waves/block
  const int GB128 = (NTOT / 16) * 1 / 8;  // Nc=128 -> 512 blocks
  const int GB256 = (NTOT / 16) * 2 / 8;  // Nc=256 -> 1024 blocks
  const int GB384 = (NTOT / 16) * 3 / 8;  // Nc=384 -> 1536 blocks

  for (int i = 0; i < NLAY; ++i) {
    // --- GINE branch ---
    gine_agg_kernel<<<GNUM, 256, NPG * CDIM * sizeof(float), stream>>>(
        h, eatt, edge_w, edge_b, srcI, dstI, z0);
    gemm_bf16_wmma<<<GB128, 256, 0, stream>>>(z0, CDIM, g1op(i), gine_b1 + i * CDIM,
                                              nullptr, t1, CDIM, CDIM, 1);
    gemm_bf16_wmma<<<GB128, 256, 0, stream>>>(t1, CDIM, g2op(i), gine_b2 + i * CDIM,
                                              h, r1, CDIM, CDIM, 0);
    run_bn(r1, g1 + i * CDIM, be1 + i * CDIM, nullptr, h1);

    // --- attention branch ---
    gemm_bf16_wmma<<<GB384, 256, 0, stream>>>(h, CDIM, inop(i), attn_in_b + i * 384,
                                              nullptr, qkv, 384, CDIM, 0);
    attn_kernel<<<GNUM * 4, 256, 122880, stream>>>(qkv, oa);
    gemm_bf16_wmma<<<GB128, 256, 0, stream>>>(oa, CDIM, outop(i), attn_out_b + i * CDIM,
                                              h, r2, CDIM, CDIM, 0);
    run_bn(r2, g2 + i * CDIM, be2 + i * CDIM, h1, o12);  // o12 = h1 + h2

    // --- MLP + final BN ---
    gemm_bf16_wmma<<<GB256, 256, 0, stream>>>(o12, CDIM, m1op(i), mlp_b1 + i * 256,
                                              nullptr, t2, 256, CDIM, 1);
    gemm_bf16_wmma<<<GB128, 256, 0, stream>>>(t2, 256, m2op(i), mlp_b2 + i * CDIM,
                                              o12, r3, CDIM, 256, 0);
    run_bn(r3, g3 + i * CDIM, be3 + i * CDIM, nullptr, h);  // h for next layer
  }

  pool_linear_kernel<<<GNUM, 128, 0, stream>>>(h, lin_w, lin_b, (float*)d_out);
}
